// Global_Attention_80393197846864
// MI455X (gfx1250) — matrix operand order repaired and checked by reading, not compiled, not run on hardware
//
#include <hip/hip_runtime.h>

typedef __attribute__((ext_vector_type(16))) __bf16 v16bf;
typedef __attribute__((ext_vector_type(8)))  float  v8f;
typedef __attribute__((ext_vector_type(4)))  float  v4f;

#define B_   16
#define N_   128
#define H_   256
#define ROWS (B_*N_*N_)     // 262144 rows of bin_features / atom_pair
#define TILES (ROWS/16)     // 16384 M-tiles of 16 rows
#define GRID_MAIN 2048

// ---------------- prep: xw = inputs @ W_atom (tiny f32 GEMM) ----------------
__global__ void prep_xw_kernel(const float* __restrict__ inputs,
                               const float* __restrict__ W_atom,
                               float* __restrict__ xw) {
  __shared__ float sin_[H_];
  const int row = blockIdx.x;          // 0 .. B*N-1
  const int t   = threadIdx.x;         // 0 .. 255
  sin_[t] = inputs[(size_t)row * H_ + t];
  __syncthreads();
  float s = 0.f;
#pragma unroll 8
  for (int k = 0; k < H_; ++k) s += sin_[k] * W_atom[k * H_ + t];
  xw[(size_t)row * H_ + t] = s;
}

// ---- prep: pack W_bin into bf16 WMMA B-fragment layout; zero S accumulator ----
// B fragment (32x16, 16-bit): lane L holds col n=(L&15); K = 8*(L>>4) + (e&7) + 16*(e>>3), e=0..15.
// packed index o = (((kstep*16 + ntile)*32 + L)*16 + e)
__global__ void prep_pack_kernel(const float* __restrict__ W_bin,
                                 unsigned short* __restrict__ packed,
                                 float* __restrict__ S) {
  const int o   = blockIdx.x * 256 + threadIdx.x;   // 0..65535
  const int e   = o & 15;
  const int L   = (o >> 4) & 31;
  const int gnt = (o >> 9) & 15;
  const int ks  = o >> 13;
  const int k   = 32 * ks + 8 * (L >> 4) + (e & 7) + 16 * (e >> 3);      // K index (h of bin)
  const int n   = (L & 15) + 16 * gnt;              // N index (output h)
  const __bf16 v = (__bf16)W_bin[k * H_ + n];
  packed[o] = __builtin_bit_cast(unsigned short, v);
  if (o < B_ * N_) S[o] = 0.f;
}

// ---------------- main fused kernel ----------------
__global__ __launch_bounds__(256, 1)
void main_kernel(const float* __restrict__ inputs,
                 const float* __restrict__ bin,        // [ROWS][H] f32
                 const float* __restrict__ xw,         // [B*N][H]  f32
                 const unsigned short* __restrict__ packed, // bf16 B frags
                 const float* __restrict__ b_bin,
                 const float* __restrict__ w_score,
                 const float* __restrict__ b_score,
                 float* __restrict__ S,                // [B*N] score sums
                 float* __restrict__ atom_pair)        // [ROWS][H] f32
{
  const int tid = threadIdx.x;
  const int wv  = tid >> 5;        // wave 0..7 -> columns 32*wv .. 32*wv+31
  const int L   = tid & 31;
  const int Lhi = L >> 4;          // 0/1
  const int Llo = L & 15;

  // This wave's B fragments: 8 ksteps x 2 ntiles, kept in VGPRs for all tiles.
  v16bf bf[16];
  {
    const v16bf* pv = (const v16bf*)packed;
#pragma unroll
    for (int ks = 0; ks < 8; ++ks)
#pragma unroll
      for (int nt = 0; nt < 2; ++nt)
        bf[ks * 2 + nt] = pv[(ks * 16 + (2 * wv + nt)) * 32 + L];
  }

  __shared__ float s_score[16];
  const float bsc = b_score[0];

  for (int tile = blockIdx.x; tile < TILES; tile += GRID_MAIN) {
    const int jt = tile & 7;
    const int i  = (tile >> 3) & (N_ - 1);
    const int b  = tile >> 10;
    const int j0 = jt * 16;
    const int bi = b * N_ + i;
    const long row0 = (long)bi * N_ + j0;   // first flattened (b,i,j) row of tile

    if (tid < 16) s_score[tid] = 0.f;

    // ---- atom_pair[b,i,j0+r,:] = inputs[b,i,:] + inputs[b,j0+r,:] (16x256 f32) ----
    // Streaming output, never re-read: non-temporal stores keep it out of L2.
    {
      const int r   = tid >> 4;       // row 0..15
      const int f4q = tid & 15;
      const v4f* in4 = (const v4f*)inputs;
      v4f* out4 = (v4f*)atom_pair;
      const long ibase = (long)bi * (H_ / 4);
      const long jbase = (long)(b * N_ + j0 + r) * (H_ / 4);
      const long obase = (row0 + r) * (H_ / 4);
#pragma unroll
      for (int q = 0; q < 4; ++q) {
        const int f4 = f4q + 16 * q;
        const v4f a = in4[ibase + f4];
        const v4f c = in4[jbase + f4];
        const v4f o = a + c;
        __builtin_nontemporal_store(o, &out4[obase + f4]);
      }
    }

    __syncthreads();

    // ---- GEMM: binW = bin[16 rows] @ W_bin[:, this wave's 32 cols], f32 accum ----
    // bin_features is streamed exactly once (256MB > L2): non-temporal loads.
    v8f acc0 = {}; v8f acc1 = {};
    const v4f* bin4 = (const v4f*)bin + (row0 + Llo) * (H_ / 4);
#pragma unroll
    for (int ks = 0; ks < 8; ++ks) {
      // 16-bit A layout: lane L, elems 0..7 -> K = 32*ks + 8*Lhi + 0..7,
      //                          elems 8..15 -> K = 32*ks + 8*Lhi + 16..23
      const int k4 = 8 * ks + 2 * Lhi;
      const v4f c0a = __builtin_nontemporal_load(&bin4[k4]);
      const v4f c0b = __builtin_nontemporal_load(&bin4[k4 + 1]);
      const v4f c1a = __builtin_nontemporal_load(&bin4[k4 + 4]);
      const v4f c1b = __builtin_nontemporal_load(&bin4[k4 + 5]);
      v16bf a;
      a[0]=(__bf16)c0a.x;  a[1]=(__bf16)c0a.y;  a[2]=(__bf16)c0a.z;  a[3]=(__bf16)c0a.w;
      a[4]=(__bf16)c0b.x;  a[5]=(__bf16)c0b.y;  a[6]=(__bf16)c0b.z;  a[7]=(__bf16)c0b.w;
      a[8]=(__bf16)c1a.x;  a[9]=(__bf16)c1a.y;  a[10]=(__bf16)c1a.z; a[11]=(__bf16)c1a.w;
      a[12]=(__bf16)c1b.x; a[13]=(__bf16)c1b.y; a[14]=(__bf16)c1b.z; a[15]=(__bf16)c1b.w;
      acc0 = __builtin_amdgcn_wmma_f32_16x16x32_bf16(false, a, false, bf[ks*2+0],
                                                     (short)0, acc0, false, false);
      acc1 = __builtin_amdgcn_wmma_f32_16x16x32_bf16(false, a, false, bf[ks*2+1],
                                                     (short)0, acc1, false, false);
    }

    // ---- epilogue: hidden = acc + xw_i + xw_j + b_bin; relu; dot w_score ----
    // C/D layout: lane L, reg r -> row M = r + 8*Lhi, col N = Llo
    float p[8];
#pragma unroll
    for (int r = 0; r < 8; ++r) p[r] = 0.f;
#pragma unroll
    for (int nt = 0; nt < 2; ++nt) {
      const int h = Llo + 32 * wv + 16 * nt;
      const float base = xw[(long)bi * H_ + h] + b_bin[h];
      const float wsc  = w_score[h];
      const v8f acc = nt ? acc1 : acc0;
#pragma unroll
      for (int r = 0; r < 8; ++r) {
        const int j = j0 + r + 8 * Lhi;
        const float xwj = xw[(long)(b * N_ + j) * H_ + h];
        float hid = acc[r] + base + xwj;
        hid = fmaxf(hid, 0.f);
        p[r] += hid * wsc;
      }
    }
    // reduce over the 16 lanes sharing a row (each half-wave independently)
#pragma unroll
    for (int r = 0; r < 8; ++r) {
      p[r] += __shfl_xor(p[r], 1, 16);
      p[r] += __shfl_xor(p[r], 2, 16);
      p[r] += __shfl_xor(p[r], 4, 16);
      p[r] += __shfl_xor(p[r], 8, 16);
    }
    if (Llo == 0) {
#pragma unroll
      for (int r = 0; r < 8; ++r)
        atomicAdd(&s_score[r + 8 * Lhi], p[r]);
    }
    __syncthreads();
    if (tid < 16) atomicAdd(&S[bi], s_score[tid] + bsc);
  }
}

// ---------------- finalize: att_context = inputs * S[b,i] ----------------
__global__ void finalize_kernel(const float* __restrict__ inputs,
                                const float* __restrict__ S,
                                float* __restrict__ out) {
  const int row = blockIdx.x;
  const int t   = threadIdx.x;
  out[(size_t)row * H_ + t] = inputs[(size_t)row * H_ + t] * S[row];
}

extern "C" void kernel_launch(void* const* d_in, const int* in_sizes, int n_in,
                              void* d_out, int out_size, void* d_ws, size_t ws_size,
                              hipStream_t stream) {
  const float* inputs  = (const float*)d_in[0];
  const float* bin     = (const float*)d_in[1];
  const float* W_atom  = (const float*)d_in[2];
  const float* W_bin   = (const float*)d_in[3];
  const float* b_bin   = (const float*)d_in[4];
  const float* w_score = (const float*)d_in[5];
  const float* b_score = (const float*)d_in[6];

  float* att_context = (float*)d_out;                           // B*N*H
  float* atom_pair   = (float*)d_out + (size_t)B_ * N_ * H_;    // B*N*N*H

  char* ws = (char*)d_ws;
  float* xw = (float*)ws;                                          // 2 MB
  unsigned short* packed = (unsigned short*)(ws + (size_t)B_*N_*H_*4); // 128 KB
  float* S = (float*)(ws + (size_t)B_*N_*H_*4 + (size_t)65536*2);  // 8 KB

  prep_xw_kernel  <<<B_*N_, H_, 0, stream>>>(inputs, W_atom, xw);
  prep_pack_kernel<<<256,   256, 0, stream>>>(W_bin, packed, S);
  main_kernel     <<<GRID_MAIN, 256, 0, stream>>>(inputs, bin, xw, packed,
                                                  b_bin, w_score, b_score,
                                                  S, atom_pair);
  finalize_kernel <<<B_*N_, H_, 0, stream>>>(inputs, S, att_context);
}
